// GraphAttentionNetwork_27341761806470
// MI455X (gfx1250) — matrix operand order repaired and checked by reading, not compiled, not run on hardware
//
#include <hip/hip_runtime.h>
#include <hip/hip_bf16.h>

#define N_NODES 50000
#define N_EDGES 800000
#define F_IN    128
#define NCOLS   1088          // 256 (W_pre) + 768 (3 layers) + 48 (score cols) + 16 pad
#define READOUT 35

typedef __bf16 bf16_t;
typedef __attribute__((ext_vector_type(16))) __bf16 v16bf;
typedef __attribute__((ext_vector_type(8)))  float  v8f;

// ---------------------------------------------------------------- helpers
__device__ inline float leaky_clip_exp(float t) {
    t = t > 0.f ? t : 0.2f * t;          // leaky_relu(0.2)
    t = fminf(fmaxf(t, -2.f), 2.f);      // clip
    return __expf(t);
}
__device__ inline float selu_f(float x) {
    const float sc = 1.0507009873554805f, al = 1.6732632423543772f;
    return x > 0.f ? sc * x : sc * al * (__expf(x) - 1.f);
}

// ---------------------------------------------------------------- prep
__global__ void prep_a_kernel(const float* __restrict__ src, bf16_t* __restrict__ dst) {
    int i = blockIdx.x * blockDim.x + threadIdx.x;
    if (i < N_NODES * F_IN) dst[i] = (bf16_t)src[i];
}

// Pack B^T: B16[c][f], c in [0,1088)
__global__ void prep_b_kernel(const float* __restrict__ Wpre,
                              const float* __restrict__ Ak,    // (3,8,128,32)
                              const float* __restrict__ Aa,    // (3,8,64)
                              bf16_t* __restrict__ B16) {
    int i = blockIdx.x * blockDim.x + threadIdx.x;
    if (i >= NCOLS * F_IN) return;
    int c = i >> 7, f = i & 127;
    float v = 0.f;
    if (c < 256) {
        v = Wpre[f * 256 + c];                          // W_pre[f][c]
    } else if (c < 1024) {
        int j = c - 256, l = j >> 8, cc = j & 255, h = cc >> 5, u = cc & 31;
        v = Ak[((l * 8 + h) * F_IN + f) * 32 + u];      // K[l][h][f][u]
    } else if (c < 1072) {
        int j = c - 1024, l = j >> 4, r = j & 15, h = r >> 1, sd = r & 1;
        const float* kk = &Ak[((l * 8 + h) * F_IN + f) * 32];
        const float* aa = &Aa[(l * 8 + h) * 64 + sd * 32];
        float acc = 0.f;
        #pragma unroll
        for (int u = 0; u < 32; ++u) acc += kk[u] * aa[u];
        v = acc;                                        // folded K@a score column
    }
    B16[i] = (bf16_t)v;
}

// ---------------------------------------------------------------- fused WMMA GEMM
// C[50000 x 1088] = A[50000 x 128] @ B[128 x 1088], bf16 in, f32 acc.
// Block: 256 threads (8 waves) -> 128x64 tile; each wave owns a 32x32 subtile
// (2x2 of 16x16) -> 16 v_wmma per block-wave with 2x fragment reuse.
__global__ __launch_bounds__(256) void gemm_wmma_kernel(
    const bf16_t* __restrict__ A16, const bf16_t* __restrict__ B16,
    float* __restrict__ C)           // [50000][1088] raw f32
{
    __shared__ bf16_t As[128 * F_IN];   // 32 KB
    __shared__ bf16_t Bs[64 * F_IN];    // 16 KB
    const int tid  = threadIdx.x;
    const int row0 = blockIdx.x * 128;
    const int col0 = blockIdx.y * 64;

    { // cooperative tile loads (16B per thread per iter, coalesced)
        const uint4* gA = (const uint4*)A16;   // 16 uint4 per 128-elem row
        uint4* sA = (uint4*)As;
        #pragma unroll
        for (int i = tid; i < 128 * 16; i += 256) {
            int r = i >> 4, w = i & 15, gr = row0 + r;
            uint4 val = make_uint4(0u, 0u, 0u, 0u);
            if (gr < N_NODES) val = gA[gr * 16 + w];
            sA[i] = val;
        }
        const uint4* gB = (const uint4*)B16;
        uint4* sB = (uint4*)Bs;
        #pragma unroll
        for (int i = tid; i < 64 * 16; i += 256) {
            int r = i >> 4, w = i & 15;
            sB[i] = gB[(col0 + r) * 16 + w];
        }
    }
    __syncthreads();

    const int wave = tid >> 5, lane = tid & 31;
    const int wm = wave & 3, wn = wave >> 2;      // 4 M-waves x 2 N-waves
    const int lr = lane & 15;
    const bool hi = lane >= 16;
    const int ka = hi ? 8 : 0;                    // A frag: K-halves per ISA layout
    const int kb = hi ? 8 : 0;                    // B frag: K-halves per ISA layout
    const int ar0 = (wm * 32 + lr) * F_IN;        // A rows, subtile 0/1
    const int ar1 = ar0 + 16 * F_IN;
    const int bc0 = (wn * 32 + lr) * F_IN;        // B cols, subtile 0/1
    const int bc1 = bc0 + 16 * F_IN;

    v8f a00 = {0,0,0,0,0,0,0,0}, a01 = a00, a10 = a00, a11 = a00;
    #pragma unroll
    for (int ks = 0; ks < F_IN; ks += 32) {
        union { v16bf v; uint4 u[2]; } f0, f1, g0, g1;
        f0.u[0] = *(const uint4*)&As[ar0 + ks + ka];
        f0.u[1] = *(const uint4*)&As[ar0 + ks + 16 + ka];
        f1.u[0] = *(const uint4*)&As[ar1 + ks + ka];
        f1.u[1] = *(const uint4*)&As[ar1 + ks + 16 + ka];
        g0.u[0] = *(const uint4*)&Bs[bc0 + ks + kb];
        g0.u[1] = *(const uint4*)&Bs[bc0 + ks + kb + 16];
        g1.u[0] = *(const uint4*)&Bs[bc1 + ks + kb];
        g1.u[1] = *(const uint4*)&Bs[bc1 + ks + kb + 16];
        a00 = __builtin_amdgcn_wmma_f32_16x16x32_bf16(false, f0.v, false, g0.v, (short)0, a00, false, false);
        a01 = __builtin_amdgcn_wmma_f32_16x16x32_bf16(false, f0.v, false, g1.v, (short)0, a01, false, false);
        a10 = __builtin_amdgcn_wmma_f32_16x16x32_bf16(false, f1.v, false, g0.v, (short)0, a10, false, false);
        a11 = __builtin_amdgcn_wmma_f32_16x16x32_bf16(false, f1.v, false, g1.v, (short)0, a11, false, false);
    }

    // epilogue: raw store; C/D layout -> row = base + (hi?8:0) + v, col = lane%16
    const int rb = row0 + wm * 32 + (hi ? 8 : 0);
    const int cb = col0 + wn * 32 + lr;
    #pragma unroll
    for (int v = 0; v < 8; ++v) {
        int r0 = rb + v, r1 = r0 + 16;
        if (r0 < N_NODES) {
            C[(size_t)r0 * NCOLS + cb]      = a00[v];
            C[(size_t)r0 * NCOLS + cb + 16] = a01[v];
        }
        if (r1 < N_NODES) {
            C[(size_t)r1 * NCOLS + cb]      = a10[v];
            C[(size_t)r1 * NCOLS + cb + 16] = a11[v];
        }
    }
}

// ---------------------------------------------------------------- GEMM post-processing
__global__ void x_init_kernel(const float* __restrict__ C, const float* __restrict__ b_pre,
                              float* __restrict__ X) {
    int i = blockIdx.x * blockDim.x + threadIdx.x;
    if (i < N_NODES * 256) {
        int n = i >> 8, c = i & 255;
        float y = C[(size_t)n * NCOLS + c] + b_pre[c];
        X[i] = y > 0.f ? y : 0.f;
    }
}

__global__ void sd_extract_kernel(const float* __restrict__ C,
                                  float* __restrict__ SN, float* __restrict__ DN) {
    int i = blockIdx.x * blockDim.x + threadIdx.x;
    if (i >= N_NODES * 48) return;
    int n = i / 48, j = i % 48;
    int l = j >> 4, r = j & 15, h = r >> 1;
    float v = C[(size_t)n * NCOLS + 1024 + j];
    float* dst = (r & 1) ? DN : SN;
    dst[((size_t)l * N_NODES + n) * 8 + h] = v;
}

// ---------------------------------------------------------------- edge phase
__global__ void edge_denom_kernel(const int* __restrict__ first, const int* __restrict__ second,
                                  const float* __restrict__ SN, const float* __restrict__ DN,
                                  float* __restrict__ denom) {
    int e = blockIdx.x * blockDim.x + threadIdx.x;
    if (e >= N_EDGES) return;
    int s = first[e], d = second[e];
    #pragma unroll
    for (int h = 0; h < 8; ++h) {
        float sc = leaky_clip_exp(SN[s * 8 + h] + DN[d * 8 + h]);
        unsafeAtomicAdd(&denom[s * 8 + h], sc);
    }
}

// one wave per edge (8 edges / 256-thread block), no barriers:
// lane l covers channels 8l..8l+7 -> its head is l>>2, softmax weight computed locally.
__global__ __launch_bounds__(256) void edge_agg_kernel(
    const int* __restrict__ first, const int* __restrict__ second,
    const float* __restrict__ SN, const float* __restrict__ DN,
    const float* __restrict__ denom,
    const float* __restrict__ NTl,     // C + 256 + l*256 (row stride NCOLS)
    float* __restrict__ OUT) {
    int wave = threadIdx.x >> 5, lane = threadIdx.x & 31;
    int e = blockIdx.x * 8 + wave;
    int s = first[e], d = second[e];
    int h = lane >> 2;
    float sc   = leaky_clip_exp(SN[s * 8 + h] + DN[d * 8 + h]);
    float norm = sc / denom[s * 8 + h];
    const float4* p = (const float4*)&NTl[(size_t)d * NCOLS + lane * 8];
    float4 v0 = p[0], v1 = p[1];
    float* o = &OUT[s * 256 + lane * 8];
    unsafeAtomicAdd(o + 0, v0.x * norm);
    unsafeAtomicAdd(o + 1, v0.y * norm);
    unsafeAtomicAdd(o + 2, v0.z * norm);
    unsafeAtomicAdd(o + 3, v0.w * norm);
    unsafeAtomicAdd(o + 4, v1.x * norm);
    unsafeAtomicAdd(o + 5, v1.y * norm);
    unsafeAtomicAdd(o + 6, v1.z * norm);
    unsafeAtomicAdd(o + 7, v1.w * norm);
}

__global__ void node_accum_kernel(const float* __restrict__ OUT, float* __restrict__ X) {
    int i = blockIdx.x * blockDim.x + threadIdx.x;
    if (i < N_NODES * 256) {
        float v = OUT[i];
        X[i] += v > 0.f ? v : 0.f;
    }
}

// ---------------------------------------------------------------- readout
__global__ void pool_kernel(const float* __restrict__ X, const int* __restrict__ gid,
                            float* __restrict__ pooled) {
    int i = blockIdx.x * blockDim.x + threadIdx.x;
    if (i < N_NODES * 256) {
        int n = i >> 8, c = i & 255;
        unsafeAtomicAdd(&pooled[gid[n] * 256 + c], X[i]);
    }
}

__global__ __launch_bounds__(64) void mlp_kernel(
    const float* __restrict__ pooled,
    const float* __restrict__ W1, const float* __restrict__ b1,
    const float* __restrict__ W2, const float* __restrict__ b2,
    const float* __restrict__ W3, const float* __restrict__ b3,
    float* __restrict__ out) {
    __shared__ float h1[READOUT];
    __shared__ float h2[READOUT];
    int g = blockIdx.x, t = threadIdx.x;
    const float* row = &pooled[g * 256];
    if (t < READOUT) {
        float acc = b1[t];
        for (int k = 0; k < 256; ++k) acc += row[k] * W1[k * READOUT + t];
        h1[t] = selu_f(acc);
    }
    __syncthreads();
    if (t < READOUT) {
        float acc = b2[t];
        for (int k = 0; k < READOUT; ++k) acc += h1[k] * W2[k * READOUT + t];
        h2[t] = selu_f(acc);
    }
    __syncthreads();
    if (t == 0) {
        float acc = b3[0];
        for (int k = 0; k < READOUT; ++k) acc += h2[k] * W3[k];
        out[g] = acc;
    }
}

// ---------------------------------------------------------------- launcher
extern "C" void kernel_launch(void* const* d_in, const int* in_sizes, int n_in,
                              void* d_out, int out_size, void* d_ws, size_t ws_size,
                              hipStream_t stream) {
    const float* states_action = (const float*)d_in[0];
    const int*   first         = (const int*)d_in[1];
    const int*   second        = (const int*)d_in[2];
    const int*   gids          = (const int*)d_in[3];
    const float* W_pre         = (const float*)d_in[4];
    const float* b_pre         = (const float*)d_in[5];
    const float* attn_kernel   = (const float*)d_in[6];
    const float* attn_a        = (const float*)d_in[7];
    const float* W1 = (const float*)d_in[8];
    const float* b1 = (const float*)d_in[9];
    const float* W2 = (const float*)d_in[10];
    const float* b2 = (const float*)d_in[11];
    const float* W3 = (const float*)d_in[12];
    const float* b3 = (const float*)d_in[13];
    float* out = (float*)d_out;

    // workspace carve-up (256B aligned)
    char* base = (char*)d_ws;
    size_t off = 0;
    auto alloc = [&](size_t bytes) {
        void* p = base + off;
        off += (bytes + 255) & ~(size_t)255;
        return p;
    };
    float*  C      = (float*)alloc((size_t)N_NODES * NCOLS * 4); // 217.6 MB
    float*  X      = (float*)alloc((size_t)N_NODES * 256 * 4);   // 51.2 MB
    float*  OUT    = (float*)alloc((size_t)N_NODES * 256 * 4);   // 51.2 MB
    bf16_t* A16    = (bf16_t*)alloc((size_t)N_NODES * F_IN * 2); // 12.8 MB
    bf16_t* B16    = (bf16_t*)alloc((size_t)NCOLS * F_IN * 2);
    float*  SN     = (float*)alloc((size_t)3 * N_NODES * 8 * 4);
    float*  DN     = (float*)alloc((size_t)3 * N_NODES * 8 * 4);
    float*  denom  = (float*)alloc((size_t)N_NODES * 8 * 4);
    float*  pooled = (float*)alloc((size_t)256 * 256 * 4);

    // 1) precision prep
    prep_a_kernel<<<(N_NODES * F_IN + 255) / 256, 256, 0, stream>>>(states_action, A16);
    prep_b_kernel<<<(NCOLS * F_IN + 255) / 256, 256, 0, stream>>>(W_pre, attn_kernel, attn_a, B16);

    // 2) fused WMMA GEMM (x-init, all nt's, all attention score dots) + routing
    dim3 ggrid((N_NODES + 127) / 128, NCOLS / 64);
    gemm_wmma_kernel<<<ggrid, 256, 0, stream>>>(A16, B16, C);
    x_init_kernel<<<(N_NODES * 256 + 255) / 256, 256, 0, stream>>>(C, b_pre, X);
    sd_extract_kernel<<<(N_NODES * 48 + 255) / 256, 256, 0, stream>>>(C, SN, DN);

    // 3) GAT edge phase, L2-resident gather/scatter
    for (int l = 0; l < 3; ++l) {
        const float* SNl = SN + (size_t)l * N_NODES * 8;
        const float* DNl = DN + (size_t)l * N_NODES * 8;
        hipMemsetAsync(denom, 0, (size_t)N_NODES * 8 * 4, stream);
        hipMemsetAsync(OUT,   0, (size_t)N_NODES * 256 * 4, stream);
        edge_denom_kernel<<<(N_EDGES + 255) / 256, 256, 0, stream>>>(first, second, SNl, DNl, denom);
        edge_agg_kernel<<<N_EDGES / 8, 256, 0, stream>>>(first, second, SNl, DNl, denom,
                                                         C + 256 + (size_t)l * 256, OUT);
        node_accum_kernel<<<(N_NODES * 256 + 255) / 256, 256, 0, stream>>>(OUT, X);
    }

    // 4) readout
    hipMemsetAsync(pooled, 0, (size_t)256 * 256 * 4, stream);
    pool_kernel<<<(N_NODES * 256 + 255) / 256, 256, 0, stream>>>(X, gids, pooled);
    mlp_kernel<<<256, 64, 0, stream>>>(pooled, W1, b1, W2, b2, W3, b3, out);
    (void)in_sizes; (void)n_in; (void)out_size; (void)ws_size;
}
